// SelfAttention_87643102642697
// MI455X (gfx1250) — matrix operand order repaired and checked by reading, not compiled, not run on hardware
//
#include <hip/hip_runtime.h>
#include <hip/hip_bf16.h>

// ---------------------------------------------------------------------------
// Causal multi-head self-attention forward, CDNA5 (gfx1250), wave32 WMMA.
// B=2, S=2048, E=1024, H=16, Dh=64.
//
// Pipeline:
//   1) f32 -> f16 conversion of x, w_qkv, w_out
//   2) QKV GEMM (WMMA f16->f32), 32x64 tile per wave, epilogue writes
//      Q*0.125, K as [b][h][s][d], V transposed as [b][h][d][s]  (all f16)
//   3) Flash-style attention: one wave per 16-query tile, online softmax,
//      score + accumulate WMMAs, causal-truncated key loop
//   4) Output projection GEMM (WMMA, 32x64 tile), f32 + bias -> d_out
// ---------------------------------------------------------------------------

typedef __attribute__((ext_vector_type(16))) _Float16 v16h;
typedef __attribute__((ext_vector_type(8)))  _Float16 v8h;
typedef __attribute__((ext_vector_type(4)))  _Float16 v4h;
typedef __attribute__((ext_vector_type(8)))  float    v8f;

#define BATCH   2
#define SEQ     2048
#define EMB     1024
#define HEADS   16
#define DHEAD   64
#define NQKV    3072

__device__ __forceinline__ v8f wmma_f16(v16h a, v16h b, v8f c) {
  // D = A(16x32 f16) * B(32x16 f16) + C(16x16 f32)
  return __builtin_amdgcn_wmma_f32_16x16x32_f16(
      /*neg_a=*/false, a, /*neg_b=*/false, b,
      /*c_mod=*/(short)0, c, /*reuse_a=*/false, /*reuse_b=*/false);
}

// A-matrix 16x32 f16 tile from row-major [rows, ld]:
// lane r=lane&15 holds row (row+r); half h=lane>>4 holds K in
// {h*8..h*8+7} (VGPR0-3) and {16+h*8..16+h*8+7} (VGPR4-7).   (ISA 7.12.2)
__device__ __forceinline__ v16h load_tileA(const _Float16* __restrict__ base,
                                           int ld, int row, int k0, int lane) {
  const int r = lane & 15, h = lane >> 4;
  const _Float16* p = base + (size_t)(row + r) * ld + k0 + h * 8;
  const v8h lo = *(const v8h*)p;
  const v8h hi = *(const v8h*)(p + 16);
  v16h out;
#pragma unroll
  for (int i = 0; i < 8; ++i) { out[i] = lo[i]; out[i + 8] = hi[i]; }
  return out;
}

// B-matrix 32x16 f16 tile, B[k][n]: lane n=lane&15 holds column (row+n) of the
// row-major source (source row = n, contiguous K); half h holds K in
// {h*8..h*8+7} (VGPR0-3) and {16+h*8..16+h*8+7} (VGPR4-7), matching A.
__device__ __forceinline__ v16h load_tileB(const _Float16* __restrict__ base,
                                           int ld, int row, int k0, int lane) {
  const int r = lane & 15, h = lane >> 4;
  const _Float16* p = base + (size_t)(row + r) * ld + k0 + h * 8;
  const v8h lo = *(const v8h*)p;
  const v8h hi = *(const v8h*)(p + 16);
  v16h out;
#pragma unroll
  for (int i = 0; i < 8; ++i) { out[i] = lo[i]; out[i + 8] = hi[i]; }
  return out;
}

// ---------------------------------------------------------------------------
// 1) f32 -> f16 conversion, vectorized x4
// ---------------------------------------------------------------------------
__global__ void cvt_f32_to_f16(const float* __restrict__ in,
                               _Float16* __restrict__ out, int n4) {
  int i = blockIdx.x * blockDim.x + threadIdx.x;
  if (i < n4) {
    const float4 f = ((const float4*)in)[i];
    v4h o;
    o[0] = (_Float16)f.x; o[1] = (_Float16)f.y;
    o[2] = (_Float16)f.z; o[3] = (_Float16)f.w;
    ((v4h*)out)[i] = o;
  }
}

// ---------------------------------------------------------------------------
// 2) QKV projection: C[4096,3072] = Xh[4096,1024] @ Wh[3072,1024]^T + bias
//    grid(48, 16), block 256: each wave computes 32 rows x 64 cols
//    (8 WMMAs per k-step, B tiles reused across both M halves).
//    n0 is 64-aligned -> one (three, head) per wave tile.
// ---------------------------------------------------------------------------
__global__ __launch_bounds__(256)
void qkv_gemm_kernel(const _Float16* __restrict__ xh,
                     const _Float16* __restrict__ wh,
                     const float* __restrict__ bias,
                     _Float16* __restrict__ qh,
                     _Float16* __restrict__ kh,
                     _Float16* __restrict__ vt) {
  const int lane = threadIdx.x & 31;
  const int wave = threadIdx.x >> 5;
  const int m0 = (blockIdx.y * 8 + wave) * 32;   // row in [0,4096), 32 rows
  const int n0 = blockIdx.x * 64;                // col in [0,3072)

  v8f c00 = {}, c01 = {}, c02 = {}, c03 = {};    // rows m0..m0+15
  v8f c10 = {}, c11 = {}, c12 = {}, c13 = {};    // rows m0+16..m0+31
  for (int k0 = 0; k0 < EMB; k0 += 32) {
    const v16h a0 = load_tileA(xh, EMB, m0,      k0, lane);
    const v16h a1 = load_tileA(xh, EMB, m0 + 16, k0, lane);
    const v16h b0 = load_tileB(wh, EMB, n0 +  0, k0, lane);
    const v16h b1 = load_tileB(wh, EMB, n0 + 16, k0, lane);
    const v16h b2 = load_tileB(wh, EMB, n0 + 32, k0, lane);
    const v16h b3 = load_tileB(wh, EMB, n0 + 48, k0, lane);
    c00 = wmma_f16(a0, b0, c00);  c10 = wmma_f16(a1, b0, c10);
    c01 = wmma_f16(a0, b1, c01);  c11 = wmma_f16(a1, b1, c11);
    c02 = wmma_f16(a0, b2, c02);  c12 = wmma_f16(a1, b2, c12);
    c03 = wmma_f16(a0, b3, c03);  c13 = wmma_f16(a1, b3, c13);
  }

  const int nlo = lane & 15, h = lane >> 4;
  const int three = n0 / EMB;            // 0=Q, 1=K, 2=V
  const int head  = (n0 % EMB) / DHEAD;
  const int bidx  = m0 / SEQ;            // 32 rows never straddle batches
  const int sbase = m0 % SEQ;
  v8f cc[2][4] = {{c00, c01, c02, c03}, {c10, c11, c12, c13}};
#pragma unroll
  for (int g = 0; g < 2; ++g) {
#pragma unroll
    for (int t = 0; t < 4; ++t) {
      const int d = t * 16 + nlo;
      const float bv = bias[n0 + t * 16 + nlo];
#pragma unroll
      for (int v = 0; v < 8; ++v) {
        const int s = sbase + g * 16 + v + 8 * h;
        const float val = cc[g][t][v] + bv;
        const size_t bh = (size_t)bidx * HEADS + head;
        if (three == 0)        // Q: fold 1/sqrt(64) into Q
          qh[(bh * SEQ + s) * DHEAD + d] = (_Float16)(val * 0.125f);
        else if (three == 1)   // K: [b][h][s][d]
          kh[(bh * SEQ + s) * DHEAD + d] = (_Float16)val;
        else                   // V transposed: [b][h][d][s]
          vt[(bh * DHEAD + d) * SEQ + s] = (_Float16)val;
      }
    }
  }
}

// ---------------------------------------------------------------------------
// 3) Flash attention: one wave (single-wave workgroup) per 16-query tile.
//    grid(S/16, H, B), block 32.
// ---------------------------------------------------------------------------
__global__ __launch_bounds__(32)
void attn_kernel(const _Float16* __restrict__ qh,
                 const _Float16* __restrict__ kh,
                 const _Float16* __restrict__ vt,
                 const int* __restrict__ causal_ptr,
                 _Float16* __restrict__ attnh) {
  const int lane = threadIdx.x;
  const int q0   = blockIdx.x * 16;
  const int head = blockIdx.y;
  const int b    = blockIdx.z;
  const int causal = causal_ptr[0];

  __shared__ __align__(16) float    sS[16][32];   // raw scores
  __shared__ __align__(16) _Float16 sP[16][32];   // exp probs (f16)
  __shared__ float sCorr[16];
  __shared__ float sLinv[16];

  const size_t bh = (size_t)b * HEADS + head;
  const _Float16* Q = qh + bh * SEQ * DHEAD;
  const _Float16* K = kh + bh * SEQ * DHEAD;
  const _Float16* V = vt + bh * DHEAD * SEQ;      // [d][s]

  const v16h qa0 = load_tileA(Q, DHEAD, q0, 0,  lane);
  const v16h qa1 = load_tileA(Q, DHEAD, q0, 32, lane);

  float m_i = -1e30f, l_i = 0.0f;                 // valid in lanes 0..15
  v8f o0 = {}, o1 = {}, o2 = {}, o3 = {};
  const int nlo = lane & 15, h = lane >> 4;
  const int kend = causal ? (q0 + 16) : SEQ;

  for (int kb = 0; kb < kend; kb += 32) {
    // ---- scores: Q(16x64) @ K^T -> 16x32 (two 16x16 C tiles) ----
    const v16h k0 = load_tileB(K, DHEAD, kb,      0,  lane);
    const v16h k1 = load_tileB(K, DHEAD, kb,      32, lane);
    const v16h k2 = load_tileB(K, DHEAD, kb + 16, 0,  lane);
    const v16h k3 = load_tileB(K, DHEAD, kb + 16, 32, lane);
    v8f slo = {}; slo = wmma_f16(qa0, k0, slo); slo = wmma_f16(qa1, k1, slo);
    v8f shi = {}; shi = wmma_f16(qa0, k2, shi); shi = wmma_f16(qa1, k3, shi);

#pragma unroll
    for (int v = 0; v < 8; ++v) {
      sS[v + 8 * h][nlo]      = slo[v];
      sS[v + 8 * h][16 + nlo] = shi[v];
    }
    __syncthreads();   // single-wave WG: S_NOP at runtime, compiler fence

    // ---- online softmax, lanes 0..15 own one query row each ----
    if (lane < 16) {
      const int qg = q0 + lane;
      float mnew = m_i;
#pragma unroll
      for (int j = 0; j < 32; ++j) {
        const bool ok = (!causal) || (kb + j <= qg);
        const float sv = ok ? sS[lane][j] : -1e30f;
        mnew = fmaxf(mnew, sv);
      }
      const float corr = __expf(m_i - mnew);
      float ssum = 0.0f;
#pragma unroll
      for (int j = 0; j < 32; ++j) {
        const bool ok = (!causal) || (kb + j <= qg);
        const float p = ok ? __expf(sS[lane][j] - mnew) : 0.0f;
        sP[lane][j] = (_Float16)p;
        ssum += p;
      }
      l_i = l_i * corr + ssum;
      m_i = mnew;
      sCorr[lane] = corr;
    }
    __syncthreads();

    // ---- rescale O, then O += P(16x32) @ Vt-tiles(32x16 each) ----
    float cf[8];
#pragma unroll
    for (int v = 0; v < 8; ++v) cf[v] = sCorr[v + 8 * h];
#pragma unroll
    for (int v = 0; v < 8; ++v) {
      o0[v] *= cf[v]; o1[v] *= cf[v]; o2[v] *= cf[v]; o3[v] *= cf[v];
    }

    // repack P from LDS into A-matrix layout
    const v8h plo = *(const v8h*)&sP[nlo][h * 8];
    const v8h phi = *(const v8h*)&sP[nlo][16 + h * 8];
    v16h pa;
#pragma unroll
    for (int i = 0; i < 8; ++i) { pa[i] = plo[i]; pa[i + 8] = phi[i]; }

    const v16h vb0 = load_tileB(V, SEQ,  0, kb, lane);
    const v16h vb1 = load_tileB(V, SEQ, 16, kb, lane);
    const v16h vb2 = load_tileB(V, SEQ, 32, kb, lane);
    const v16h vb3 = load_tileB(V, SEQ, 48, kb, lane);
    o0 = wmma_f16(pa, vb0, o0);
    o1 = wmma_f16(pa, vb1, o1);
    o2 = wmma_f16(pa, vb2, o2);
    o3 = wmma_f16(pa, vb3, o3);
    __syncthreads();
  }

  if (lane < 16) sLinv[lane] = 1.0f / l_i;
  __syncthreads();
  float lf[8];
#pragma unroll
  for (int v = 0; v < 8; ++v) lf[v] = sLinv[v + 8 * h];

  // write attention output as f16 in (b, s, h*d) for the final GEMM
  _Float16* dst = attnh + ((size_t)b * SEQ + q0) * EMB + head * DHEAD;
  v8f oo[4] = {o0, o1, o2, o3};
#pragma unroll
  for (int t = 0; t < 4; ++t)
#pragma unroll
    for (int v = 0; v < 8; ++v)
      dst[(size_t)(v + 8 * h) * EMB + t * 16 + nlo] = (_Float16)(oo[t][v] * lf[v]);
}

// ---------------------------------------------------------------------------
// 4) Output projection: Y[4096,1024] = Ah @ Wout^T + b_out (f32 out)
//    grid(16, 16), block 256, 32x64 tile per wave.
// ---------------------------------------------------------------------------
__global__ __launch_bounds__(256)
void out_gemm_kernel(const _Float16* __restrict__ ah,
                     const _Float16* __restrict__ wh,
                     const float* __restrict__ bias,
                     float* __restrict__ out) {
  const int lane = threadIdx.x & 31;
  const int wave = threadIdx.x >> 5;
  const int m0 = (blockIdx.y * 8 + wave) * 32;
  const int n0 = blockIdx.x * 64;

  v8f c00 = {}, c01 = {}, c02 = {}, c03 = {};
  v8f c10 = {}, c11 = {}, c12 = {}, c13 = {};
  for (int k0 = 0; k0 < EMB; k0 += 32) {
    const v16h a0 = load_tileA(ah, EMB, m0,      k0, lane);
    const v16h a1 = load_tileA(ah, EMB, m0 + 16, k0, lane);
    const v16h b0 = load_tileB(wh, EMB, n0 +  0, k0, lane);
    const v16h b1 = load_tileB(wh, EMB, n0 + 16, k0, lane);
    const v16h b2 = load_tileB(wh, EMB, n0 + 32, k0, lane);
    const v16h b3 = load_tileB(wh, EMB, n0 + 48, k0, lane);
    c00 = wmma_f16(a0, b0, c00);  c10 = wmma_f16(a1, b0, c10);
    c01 = wmma_f16(a0, b1, c01);  c11 = wmma_f16(a1, b1, c11);
    c02 = wmma_f16(a0, b2, c02);  c12 = wmma_f16(a1, b2, c12);
    c03 = wmma_f16(a0, b3, c03);  c13 = wmma_f16(a1, b3, c13);
  }

  const int nlo = lane & 15, h = lane >> 4;
  v8f cc[2][4] = {{c00, c01, c02, c03}, {c10, c11, c12, c13}};
#pragma unroll
  for (int g = 0; g < 2; ++g) {
#pragma unroll
    for (int t = 0; t < 4; ++t) {
      const float bv = bias[n0 + t * 16 + nlo];
#pragma unroll
      for (int v = 0; v < 8; ++v)
        out[(size_t)(m0 + g * 16 + v + 8 * h) * EMB + n0 + t * 16 + nlo] =
            cc[g][t][v] + bv;
    }
  }
}

// ---------------------------------------------------------------------------
// Host launcher
// ---------------------------------------------------------------------------
extern "C" void kernel_launch(void* const* d_in, const int* in_sizes, int n_in,
                              void* d_out, int out_size, void* d_ws, size_t ws_size,
                              hipStream_t stream) {
  (void)in_sizes; (void)n_in; (void)out_size; (void)ws_size;

  const float* x      = (const float*)d_in[0];   // [2,2048,1024]
  const float* w_qkv  = (const float*)d_in[1];   // [3072,1024]
  const float* b_qkv  = (const float*)d_in[2];   // [3072]
  const float* w_out  = (const float*)d_in[3];   // [1024,1024]
  const float* b_out  = (const float*)d_in[4];   // [1024]
  const int*   causal = (const int*)d_in[5];     // scalar
  float* out = (float*)d_out;

  // Workspace carve-out (f16 buffers), 48 MB total
  char* ws = (char*)d_ws;
  _Float16* xh    = (_Float16*)(ws + 0);                       //  8 MB
  _Float16* wqkvh = (_Float16*)(ws + (8u << 20));              //  6 MB
  _Float16* wouth = (_Float16*)(ws + (14u << 20));             //  2 MB
  _Float16* qh    = (_Float16*)(ws + (16u << 20));             //  8 MB
  _Float16* kh    = (_Float16*)(ws + (24u << 20));             //  8 MB
  _Float16* vt    = (_Float16*)(ws + (32u << 20));             //  8 MB
  _Float16* attnh = (_Float16*)(ws + (40u << 20));             //  8 MB

  const int nX = BATCH * SEQ * EMB;     // 4194304
  const int nWq = NQKV * EMB;           // 3145728
  const int nWo = EMB * EMB;            // 1048576

  cvt_f32_to_f16<<<(nX / 4 + 255) / 256, 256, 0, stream>>>(x, xh, nX / 4);
  cvt_f32_to_f16<<<(nWq / 4 + 255) / 256, 256, 0, stream>>>(w_qkv, wqkvh, nWq / 4);
  cvt_f32_to_f16<<<(nWo / 4 + 255) / 256, 256, 0, stream>>>(w_out, wouth, nWo / 4);

  // QKV projection: 4096 x 3072, 32 rows per wave
  qkv_gemm_kernel<<<dim3(NQKV / 64, (BATCH * SEQ) / (32 * 8)), 256, 0, stream>>>(
      xh, wqkvh, b_qkv, qh, kh, vt);

  // Flash attention
  attn_kernel<<<dim3(SEQ / 16, HEADS, BATCH), 32, 0, stream>>>(
      qh, kh, vt, causal, attnh);

  // Output projection: 4096 x 1024, 32 rows per wave
  out_gemm_kernel<<<dim3(EMB / 64, (BATCH * SEQ) / (32 * 8)), 256, 0, stream>>>(
      attnh, wouth, b_out, out);
}
